// LIFNetwork_44856638439521
// MI455X (gfx1250) — matrix operand order repaired and checked by reading, not compiled, not run on hardware
//
#include <hip/hip_runtime.h>
#include <hip/hip_bf16.h>

// ---------- types ----------
typedef __attribute__((ext_vector_type(16))) __bf16 v16bf;
typedef __attribute__((ext_vector_type(8)))  __bf16 v8bf;
typedef __attribute__((ext_vector_type(8)))  float  v8f;

#define MAX_N 8192

__device__ __forceinline__ unsigned short f32_to_bf16_bits(float f) {
    unsigned u = __builtin_bit_cast(unsigned, f);
    // round-to-nearest-even
    unsigned r = (u + 0x7FFFu + ((u >> 16) & 1u)) >> 16;
    return (unsigned short)r;
}

// ---------- one-time: build transposed, scaled bf16 effective weights ----------
// w_eff[j,i] = W[j,i] * ((type[j]==1 ? Ew : Iw) * type[j]), stored transposed:
// wT[i*N + j], so each WMMA B-fragment lane reads 16 consecutive K values
// (32 contiguous bytes). bf16 halves the matrix to 128 MB -> resident in the
// 192 MB L2, so the 500 steps stream from L2 instead of HBM.
__global__ __launch_bounds__(256) void build_wT(const float* __restrict__ W,
                                                const int*   __restrict__ ntype,
                                                const float* __restrict__ Ew,
                                                const float* __restrict__ Iw,
                                                unsigned short* __restrict__ wT,
                                                int N) {
    __shared__ unsigned short tile[32][34];   // +2 pad: avoid bank conflicts
    const int j0 = blockIdx.y * 32;           // presynaptic rows
    const int i0 = blockIdx.x * 32;           // postsynaptic cols
    const int tx = threadIdx.x;               // 0..31
    const int ty = threadIdx.y;               // 0..7
    const float ew = *Ew, iw = *Iw;
#pragma unroll
    for (int r = 0; r < 4; ++r) {
        const int j = j0 + ty + r * 8;
        const int t = ntype[j];
        const float sc = (t == 1 ? ew : iw) * (float)t;
        const float v  = W[(size_t)j * N + i0 + tx] * sc;
        tile[ty + r * 8][tx] = f32_to_bf16_bits(v);
    }
    __syncthreads();
#pragma unroll
    for (int r = 0; r < 4; ++r) {
        const int i = i0 + ty + r * 8;
        wT[(size_t)i * N + j0 + tx] = tile[tx][ty + r * 8];
    }
}

// ---------- one-time: zero state ----------
__global__ void init_state(float* v, float* I,
                           unsigned short* sA, unsigned short* sB, int N) {
    const int i = blockIdx.x * blockDim.x + threadIdx.x;
    if (i < N) { v[i] = 0.0f; I[i] = 0.0f; sA[i] = 0; sB[i] = 0; }
}

// ---------- per-step: s@W via WMMA + fused LIF update ----------
// Block = 256 threads = 8 waves, block owns 16 output neurons. The spike
// vector is staged into LDS once per block; per WMMA the B (weight) fragment
// is 2x global b128 (VMEM pipe) and the A (spike) fragment is 2x ds b128
// (LDS pipe, conflict-free 16-lane broadcast) -> the two memory pipes issue
// concurrently and VMEM slots hit the b128 floor.
//
// A-fragment trick: only D row M=0 is consumed; rows 1..15 of A are
// don't-cares for c[0], so every lane loads the same broadcast spike slice
// unconditionally (no EXEC masking, no zero fills).
__global__ __launch_bounds__(256) void lif_step(
        const unsigned short* __restrict__ wT_u,
        const unsigned short* __restrict__ s_in_u,
        unsigned short*       __restrict__ s_out_u,
        float* __restrict__ vbuf, float* __restrict__ Ibuf,
        const float* __restrict__ inputs, const int* __restrict__ ntype,
        const float* __restrict__ Evth, const float* __restrict__ Ivth,
        float* __restrict__ vtr, float* __restrict__ str, int N) {
    __shared__ __bf16 sLDS[MAX_N];
    __shared__ float  red[128];

    // cooperative coalesced copy: spike vector (N bf16 = 16 KB) -> LDS
    {
        const uint4* g = (const uint4*)s_in_u;
        uint4*       l = (uint4*)sLDS;
        const int nvec = N >> 3;                  // 8 bf16 per uint4
        for (int idx = threadIdx.x; idx < nvec; idx += blockDim.x) l[idx] = g[idx];
    }
    __syncthreads();

    const int lane = threadIdx.x & 31;
    const int wid  = threadIdx.x >> 5;
    const int n0   = blockIdx.x * 16;
    const int Kw   = N >> 3;              // K slice per wave (8 waves)
    const int k0   = wid * Kw;

    // B fragment: this lane's output column; hi half-wave covers K+16..31.
    const int col = n0 + (lane & 15);
    const __bf16* bptr = (const __bf16*)wT_u
                       + (size_t)col * N + ((lane >> 4) << 4) + k0;
    // A fragment: broadcast spike slice from LDS; hi half-wave covers
    // K+16..31.
    const __bf16* sptr = sLDS + k0 + ((lane >> 4) << 4);

    v8f c0 = {}, c1 = {};
    const int iters = Kw >> 6;            // 64 K-values per iteration
    for (int it = 0; it < iters; ++it) {
        v16bf b0 = *(const v16bf*)(bptr);        // K 0..15 / 16..31 of chunk
        v16bf b1 = *(const v16bf*)(bptr + 32);   // K 32..47 / 48..63
        v8bf a0lo = *(const v8bf*)(sptr);
        v8bf a0hi = *(const v8bf*)(sptr + 8);
        v8bf a1lo = *(const v8bf*)(sptr + 32);
        v8bf a1hi = *(const v8bf*)(sptr + 40);

        v16bf a0 = __builtin_shufflevector(a0lo, a0hi,
                     0,1,2,3,4,5,6,7,8,9,10,11,12,13,14,15);
        v16bf a1 = __builtin_shufflevector(a1lo, a1hi,
                     0,1,2,3,4,5,6,7,8,9,10,11,12,13,14,15);

        c0 = __builtin_amdgcn_wmma_f32_16x16x32_bf16(
                 false, a0, false, b0, (short)0, c0, false, false);
        c1 = __builtin_amdgcn_wmma_f32_16x16x32_bf16(
                 false, a1, false, b1, (short)0, c1, false, false);
        bptr += 64;
        sptr += 64;
    }

    // reduce the 8 wave partials: D row M=0 sits in element 0 on lanes 0..15
    if (lane < 16) red[wid * 16 + lane] = c0[0] + c1[0];
    __syncthreads();

    if (threadIdx.x < 16) {
        const float alpha = 0.90483741803596f;   // exp(-DT/TAU_M)
        const float beta  = 0.81873075307798f;   // exp(-DT/TAU_I)
        const int   tid = threadIdx.x;
        const int   i   = n0 + tid;
        float dot = 0.0f;
#pragma unroll
        for (int w = 0; w < 8; ++w) dot += red[w * 16 + tid];

        const float Inew = beta * Ibuf[i] + dot;
        float vnew = alpha * vbuf[i] + (1.0f - alpha) * (Inew + inputs[i]);
        const float vth = (ntype[i] == 1) ? *Evth : *Ivth;
        const float spk = (vnew >= vth) ? 1.0f : 0.0f;
        if (spk > 0.0f) vnew = 0.0f;

        Ibuf[i] = Inew;
        vbuf[i] = vnew;
        s_out_u[i] = (spk > 0.0f) ? (unsigned short)0x3F80 : (unsigned short)0x0000; // bf16 1/0
        vtr[i] = vnew;
        str[i] = spk;
    }
}

// ---------- host ----------
extern "C" void kernel_launch(void* const* d_in, const int* in_sizes, int n_in,
                              void* d_out, int out_size, void* d_ws, size_t ws_size,
                              hipStream_t stream) {
    const float* inputs = (const float*)d_in[0];
    const float* W      = (const float*)d_in[1];
    const int*   ntype  = (const int*)d_in[2];
    const float* Ew     = (const float*)d_in[3];
    const float* Iw     = (const float*)d_in[4];
    const float* Evth   = (const float*)d_in[5];
    const float* Ivth   = (const float*)d_in[6];

    const int N       = in_sizes[0];              // 8192
    const int n_steps = out_size / (2 * N);       // 500

    // workspace: [ wT : N*N bf16 ][ sA : N bf16 ][ sB : N bf16 ][ v : N f32 ][ I : N f32 ]
    char* ws = (char*)d_ws;
    unsigned short* wT = (unsigned short*)ws;
    const size_t wtBytes = (size_t)N * (size_t)N * 2u;
    unsigned short* sA = (unsigned short*)(ws + wtBytes);
    unsigned short* sB = sA + N;
    float* vbuf = (float*)(sB + N);
    float* Ibuf = vbuf + N;

    float* vtr = (float*)d_out;
    float* str = vtr + (size_t)n_steps * N;

    init_state<<<(N + 255) / 256, 256, 0, stream>>>(vbuf, Ibuf, sA, sB, N);

    dim3 tb(32, 8), tg(N / 32, N / 32);
    build_wT<<<tg, tb, 0, stream>>>(W, ntype, Ew, Iw, wT, N);

    for (int t = 0; t < n_steps; ++t) {
        const unsigned short* sin  = (t & 1) ? sB : sA;
        unsigned short*       sout = (t & 1) ? sA : sB;
        lif_step<<<N / 16, 256, 0, stream>>>(
            wT, sin, sout, vbuf, Ibuf, inputs, ntype, Evth, Ivth,
            vtr + (size_t)t * N, str + (size_t)t * N, N);
    }
}
